// ParamComponentsRankPenalty_15848429322669
// MI455X (gfx1250) — matrix operand order repaired and checked by reading, not compiled, not run on hardware
//
#include <hip/hip_runtime.h>

typedef __attribute__((ext_vector_type(16))) _Float16 v16h;
typedef __attribute__((ext_vector_type(8)))  _Float16 v8h;
typedef __attribute__((ext_vector_type(8)))  float    v8f;
typedef __attribute__((ext_vector_type(4)))  int      v4i;

#define BATCH  4096
#define DIN    1024
#define DOUT   1024
#define NEXP   16
#define MDIM   512
#define TILE_B 32

typedef __attribute__((address_space(1))) v4i* as1_v4i_p;
typedef __attribute__((address_space(3))) v4i* as3_v4i_p;

#define WAIT_ASYNC(n) asm volatile("s_wait_asynccnt " #n ::: "memory")
#define WAIT_DS0()    asm volatile("s_wait_dscnt 0" ::: "memory")

// ---------------- fragment helpers ----------------

__device__ __forceinline__ v16h cat16(v8h lo, v8h hi) {
  return __builtin_shufflevector(lo, hi, 0, 1, 2, 3, 4, 5, 6, 7,
                                 8, 9, 10, 11, 12, 13, 14, 15);
}

// LDS tile [rows][32] halves, XOR-swizzled in 16B blocks (bank-conflict-free b128s)
__device__ __forceinline__ int swz32(int row, int k) {
  return row * 32 + ((((k >> 3) ^ (row >> 2)) & 3) << 3) + (k & 7);
}

// A-operand fragment (16x32 f16): lane group g holds K in {8g..8g+7} U {16+8g..23+8g}
__device__ __forceinline__ v16h fragA32(const _Float16* buf, int row, int g) {
  const int s = (row >> 2) & 3;
  v8h lo = *(const v8h*)(buf + row * 32 + (((g) ^ s) & 3) * 8);
  v8h hi = *(const v8h*)(buf + row * 32 + (((g + 2) ^ s) & 3) * 8);
  return cat16(lo, hi);
}

// B-operand fragment (32x16 f16): lane group g holds K in {8g..8g+7} U {16+8g..23+8g}
__device__ __forceinline__ v16h fragB32(const _Float16* buf, int row, int g) {
  const int s = (row >> 2) & 3;
  v8h lo = *(const v8h*)(buf + row * 32 + (((g) ^ s) & 3) * 8);
  v8h hi = *(const v8h*)(buf + row * 32 + (((g + 2) ^ s) & 3) * 8);
  return cat16(lo, hi);
}

// preS LDS tile [32][512] halves, swizzled by (row & 7) over 16B blocks
__device__ __forceinline__ int swzP(int m, int c) {
  return m * 512 + (((c >> 3) ^ (m & 7)) << 3) + (c & 7);
}

__device__ __forceinline__ v16h fragP(const _Float16* pre, int m, int k0, int g) {
  const int b0 = (k0 >> 3) + g;
  const int sm = m & 7;
  v8h lo = *(const v8h*)(pre + m * 512 + ((b0) ^ sm) * 8);
  v8h hi = *(const v8h*)(pre + m * 512 + ((b0 + 2) ^ sm) * 8);
  return cat16(lo, hi);
}

// ---------------- async global -> LDS (gfx1250) ----------------

__device__ __forceinline__ void async_cp16(const _Float16* g, _Float16* l) {
  __builtin_amdgcn_global_load_async_to_lds_b128(
      (as1_v4i_p)(uintptr_t)g, (as3_v4i_p)(unsigned)(uintptr_t)l, 0, 0);
}

// ---------------- f32 staging (fallback path + convert kernels) ----------------

// 32 rows x 256 cols f32 -> transposed+swizzled f16 (one column/thread)
__device__ __forceinline__ void stage_w_f32(const float* src, int cols,
                                            _Float16* Wbuf, int tid) {
  const int n = tid;
  const float* s = src + n;
  const int sw = (n >> 2) & 3;
#pragma unroll
  for (int blk = 0; blk < 4; ++blk) {
    union { _Float16 h[8]; uint4 u; } p;
#pragma unroll
    for (int j = 0; j < 8; ++j)
      p.h[j] = (_Float16)s[(size_t)(blk * 8 + j) * cols];
    *(uint4*)&Wbuf[n * 32 + ((blk ^ sw) << 3)] = p.u;
  }
}

// 32 rows x 128 cols variant (threads duplicate columns pairwise; benign)
__device__ __forceinline__ void stage_w128_f32(const float* src, int cols,
                                               _Float16* Wbuf, int tid) {
  const int n = tid & 127;
  const float* s = src + n;
  const int sw = (n >> 2) & 3;
#pragma unroll
  for (int blk = 0; blk < 4; ++blk) {
    union { _Float16 h[8]; uint4 u; } p;
#pragma unroll
    for (int j = 0; j < 8; ++j)
      p.h[j] = (_Float16)s[(size_t)(blk * 8 + j) * cols];
    *(uint4*)&Wbuf[n * 32 + ((blk ^ sw) << 3)] = p.u;
  }
}

// x chunk [32 rows][32 k] f32 -> swizzled f16
__device__ __forceinline__ void stage_x_f32(const float* xrow0, _Float16* Xbuf, int tid) {
  const int r = tid >> 3, c0 = (tid & 7) << 2;
  const float* s = xrow0 + (size_t)r * DIN + c0;
  union { _Float16 h[4]; uint2 u; } p;
  p.h[0] = (_Float16)s[0]; p.h[1] = (_Float16)s[1];
  p.h[2] = (_Float16)s[2]; p.h[3] = (_Float16)s[3];
  *(uint2*)&Xbuf[swz32(r, c0)] = p.u;
}

// ---------------- pre-conversion kernels (fill d_ws) ----------------

__global__ __launch_bounds__(256) void convert_x_kernel(
    const float* __restrict__ x, _Float16* __restrict__ xF) {
  const int idx = blockIdx.x * 256 + threadIdx.x;  // over BATCH*DIN/4
  const int b = idx >> 8;
  const int c0 = (idx & 255) << 2;
  const float* s = x + (size_t)b * DIN + c0;
  union { _Float16 h[4]; uint2 u; } p;
  p.h[0] = (_Float16)s[0]; p.h[1] = (_Float16)s[1];
  p.h[2] = (_Float16)s[2]; p.h[3] = (_Float16)s[3];
  _Float16* d = xF + (size_t)((b >> 5) * 32 + (c0 >> 5)) * 1024 + swz32(b & 31, c0 & 31);
  *(uint2*)d = p.u;
}

__global__ __launch_bounds__(256) void convert_w_kernel(
    const float* __restrict__ src, _Float16* __restrict__ dst, int rows, int cols) {
  __shared__ __align__(16) _Float16 T[256 * 32];
  const int tid = threadIdx.x;
  const int ncg = cols >> 8, nic = rows >> 5;
  const int id = blockIdx.x;
  const int cg = id % ncg;
  const int ic = (id / ncg) % nic;
  const int ke = id / (ncg * nic);
  const float* s = src + ((size_t)ke * rows + ic * 32) * cols + cg * 256;
  stage_w_f32(s, cols, T, tid);
  __syncthreads();
  uint4* d = (uint4*)(dst + (size_t)id * 8192);
  const uint4* t4 = (const uint4*)T;
#pragma unroll
  for (int j = 0; j < 4; ++j) d[tid + j * 256] = t4[tid + j * 256];
}

// ---------------- fused MoE kernel ----------------

template <bool PRE>
__global__ __launch_bounds__(256) void moe_fused_kernel(
    const float* __restrict__ x, const unsigned char* __restrict__ mask,
    const float* __restrict__ A, const float* __restrict__ B,
    const _Float16* __restrict__ xF, const _Float16* __restrict__ aF,
    const _Float16* __restrict__ bF, float* __restrict__ inner) {
  // 64KB pooled LDS: preS [0,16384) | pool [16384,32768)
  __shared__ __align__(16) _Float16 S[32768];
  _Float16* const preS = S;
  _Float16* const pool = S + 16384;
  // phase A: Ws bufs at pool + buf*4096, Xs bufs at pool + 8192 + buf*1024
  // phase B: Ws bufs at pool + buf*8192

  const int tid  = threadIdx.x;
  const int w    = tid >> 5;
  const int lane = tid & 31;
  const int g    = lane >> 4;
  const int mr   = lane & 15;
  const int rb   = w & 1;   // row band
  const int cg   = w >> 1;  // col group

  const int b0 = blockIdx.x * TILE_B;
  const int ke = blockIdx.y;

  // per-lane row mask (rows rb*16 + 8g + j)
  float mk[8];
#pragma unroll
  for (int j = 0; j < 8; ++j)
    mk[j] = mask[(size_t)(b0 + rb * 16 + 8 * g + j) * NEXP + ke] ? 1.0f : 0.0f;

  const float* Ak = A + (size_t)ke * DIN * MDIM;
  const float* Bk = B + (size_t)ke * MDIM * DOUT;

  // async issue: one K-chunk group (uniform 3 ops/thread; Xs duplicated by hi waves)
  auto issueA = [&](int sect, int ic, int buf) {
    const _Float16* ga = aF + (size_t)((ke * 32 + ic) * 2 + (sect >> 1)) * 8192 +
                         (sect & 1) * 4096;
    _Float16* Wb = pool + buf * 4096;
    async_cp16(ga + tid * 8, Wb + tid * 8);
    async_cp16(ga + (tid + 256) * 8, Wb + (tid + 256) * 8);
    const _Float16* gx = xF + (size_t)((b0 >> 5) * 32 + ic) * 1024;
    _Float16* Xb = pool + 8192 + buf * 1024;
    async_cp16(gx + (tid & 127) * 8, Xb + (tid & 127) * 8);
  };
  auto issueB = [&](int q, int ic, int buf) {
    const _Float16* gb = bF + (size_t)((ke * 16 + ic) * 4 + q) * 8192;
    _Float16* Wb = pool + buf * 8192;
#pragma unroll
    for (int j = 0; j < 4; ++j)
      async_cp16(gb + (tid + j * 256) * 8, Wb + (tid + j * 256) * 8);
  };

  // ---- Phase A: preS[32][512] = Xtile @ Ak, 4 sections of 128 cols ----
  for (int sect = 0; sect < 4; ++sect) {
    v8f acc[2];
#pragma unroll
    for (int t = 0; t < 2; ++t)
      acc[t] = v8f{0.f, 0.f, 0.f, 0.f, 0.f, 0.f, 0.f, 0.f};

    if constexpr (PRE) { issueA(sect, 0, 0); issueA(sect, 1, 1); }

    for (int ic = 0; ic < 32; ++ic) {
      const int cur = ic & 1;
      if constexpr (PRE) {
        if (ic < 31) WAIT_ASYNC(3); else WAIT_ASYNC(0);
        __syncthreads();
      } else {
        __syncthreads();
        stage_x_f32(x + (size_t)b0 * DIN + ic * 32, pool + 8192, tid);
        stage_w128_f32(Ak + (size_t)(ic * 32) * MDIM + sect * 128, MDIM, pool, tid);
        __syncthreads();
      }
      const _Float16* Wb = pool + (PRE ? cur * 4096 : 0);
      const _Float16* Xb = pool + 8192 + (PRE ? cur * 1024 : 0);

      v16h af = fragA32(Xb, rb * 16 + mr, g);
#pragma unroll
      for (int t = 0; t < 2; ++t) {
        v16h bf = fragB32(Wb, (cg * 2 + t) * 16 + mr, g);
        acc[t] = __builtin_amdgcn_wmma_f32_16x16x32_f16(
            false, af, false, bf, (short)0, acc[t], false, false);
      }
      __syncthreads();
      if constexpr (PRE) { if (ic + 2 < 32) issueA(sect, ic + 2, cur); }
    }
    // epilogue: mask + convert into preS
#pragma unroll
    for (int t = 0; t < 2; ++t) {
      const int c = sect * 128 + (cg * 2 + t) * 16 + mr;
#pragma unroll
      for (int j = 0; j < 8; ++j) {
        const int m = rb * 16 + 8 * g + j;
        preS[swzP(m, c)] = (_Float16)(acc[t][j] * mk[j]);
      }
    }
  }
  WAIT_DS0();
  __syncthreads();

  // ---- Phase B: inner[32][1024] = preS @ Bk, 4 sections of 256 cols ----
  for (int q = 0; q < 4; ++q) {
    v8f acc[4];
#pragma unroll
    for (int t = 0; t < 4; ++t)
      acc[t] = v8f{0.f, 0.f, 0.f, 0.f, 0.f, 0.f, 0.f, 0.f};

    if constexpr (PRE) { issueB(q, 0, 0); issueB(q, 1, 1); }

    for (int ic = 0; ic < 16; ++ic) {
      const int cur = ic & 1;
      const int k0 = ic * 32;
      if constexpr (PRE) {
        if (ic < 15) WAIT_ASYNC(4); else WAIT_ASYNC(0);
        __syncthreads();
      } else {
        __syncthreads();
        stage_w_f32(Bk + (size_t)k0 * DOUT + q * 256, DOUT, pool, tid);
        __syncthreads();
      }
      const _Float16* Wb = pool + (PRE ? cur * 8192 : 0);

      v16h af = fragP(preS, rb * 16 + mr, k0, g);
#pragma unroll
      for (int t = 0; t < 4; ++t) {
        v16h bf = fragB32(Wb, (cg * 4 + t) * 16 + mr, g);
        acc[t] = __builtin_amdgcn_wmma_f32_16x16x32_f16(
            false, af, false, bf, (short)0, acc[t], false, false);
      }
      __syncthreads();
      if constexpr (PRE) { if (ic + 2 < 16) issueB(q, ic + 2, cur); }
    }
    // epilogue: write inner_acts (masked rows already exact zeros)
#pragma unroll
    for (int t = 0; t < 4; ++t) {
      const int hc = q * 256 + (cg * 4 + t) * 16 + mr;
#pragma unroll
      for (int j = 0; j < 8; ++j) {
        const int m = rb * 16 + 8 * g + j;
        inner[((size_t)(b0 + m) * NEXP + ke) * DOUT + hc] = acc[t][j];
      }
    }
  }
}

// out[b,h] = bias[h] + sum_k inner[b,k,h]
__global__ __launch_bounds__(256) void moe_reduce_kernel(
    const float* __restrict__ inner, const float* __restrict__ bias,
    float* __restrict__ out) {
  const size_t idx = (size_t)blockIdx.x * 256 + threadIdx.x;
  const int b = (int)(idx >> 10);
  const int hc = (int)(idx & 1023);
  float s = bias[hc];
  const float* p = inner + (size_t)b * NEXP * DOUT + hc;
#pragma unroll
  for (int k = 0; k < NEXP; ++k) s += p[(size_t)k * DOUT];
  out[idx] = s;
}

extern "C" void kernel_launch(void* const* d_in, const int* in_sizes, int n_in,
                              void* d_out, int out_size, void* d_ws, size_t ws_size,
                              hipStream_t stream) {
  const float* x            = (const float*)d_in[0];
  const unsigned char* mask = (const unsigned char*)d_in[1];  // jnp bool -> 1 byte
  const float* A            = (const float*)d_in[2];
  const float* B            = (const float*)d_in[3];
  const float* bias         = (const float*)d_in[4];

  float* out   = (float*)d_out;
  float* inner = out + (size_t)BATCH * DOUT;

  const size_t XH = (size_t)BATCH * DIN;
  const size_t AH = (size_t)NEXP * DIN * MDIM;
  const size_t BH = (size_t)NEXP * MDIM * DOUT;
  const size_t NEED = (XH + AH + BH) * sizeof(_Float16);

  dim3 grid(BATCH / TILE_B, NEXP);

  if (ws_size >= NEED) {
    _Float16* xF = (_Float16*)d_ws;
    _Float16* aF = xF + XH;
    _Float16* bF = aF + AH;
    convert_x_kernel<<<(BATCH * DIN / 4) / 256, 256, 0, stream>>>(x, xF);
    convert_w_kernel<<<NEXP * (DIN / 32) * (MDIM / 256), 256, 0, stream>>>(A, aF, DIN, MDIM);
    convert_w_kernel<<<NEXP * (MDIM / 32) * (DOUT / 256), 256, 0, stream>>>(B, bF, MDIM, DOUT);
    moe_fused_kernel<true><<<grid, 256, 0, stream>>>(x, mask, A, B, xF, aF, bF, inner);
  } else {
    moe_fused_kernel<false><<<grid, 256, 0, stream>>>(x, mask, A, B,
                                                      nullptr, nullptr, nullptr, inner);
  }

  moe_reduce_kernel<<<(BATCH * DOUT) / 256, 256, 0, stream>>>(inner, bias, out);
}
